// ContinuousThoughtMachineCell_50337016709332
// MI455X (gfx1250) — matrix operand order repaired and checked by reading, not compiled, not run on hardware
//
#include <hip/hip_runtime.h>
#include <hip/hip_bf16.h>

// Shapes from reference: B=64, D=512, DIN=512, M=32, H=64, DSQ=262144.
#define BB   64
#define DD   512
#define MM   32
#define HH   64
#define DSQ  262144

typedef __attribute__((ext_vector_type(16))) _Float16 v16h;
typedef __attribute__((ext_vector_type(8)))  float    v8f;
typedef __attribute__((ext_vector_type(4)))  float    f32x4;

// ---- WMMA frag helpers (contiguous-K order, same for A and B, wave32) -------------------
// A 16x32 f16: lane L holds row M = L%16; halfs i -> K = i + (L<16?0:16)
__device__ __forceinline__ v16h frag_a_lds(const _Float16* s, int lda, int mBase,
                                           int k0, int lane) {
  v16h a;
  const _Float16* p = s + (mBase + (lane & 15)) * lda;
  int kb = k0 + ((lane & 16) ? 16 : 0);
#pragma unroll
  for (int i = 0; i < 16; ++i) a[i] = p[kb + i];
  return a;
}
// B 32x16 f16: lane L holds col N = L%16; halfs i -> K = i + (L<16?0:16)
__device__ __forceinline__ v16h frag_b_lds(const _Float16* s, int ldb, int k0,
                                           int nBase, int lane) {
  v16h b;
  int col = nBase + (lane & 15);
  int kb = k0 + ((lane & 16) ? 16 : 0);
#pragma unroll
  for (int i = 0; i < 16; ++i) b[i] = s[(kb + i) * ldb + col];
  return b;
}
__device__ __forceinline__ v8f wmma_f16(v16h a, v16h b, v8f c) {
  return __builtin_amdgcn_wmma_f32_16x16x32_f16(false, a, false, b, (short)0, c,
                                                false, false);
}

// ---- K0: invNorm = 1/||weights_fixed||_F ------------------------------------
__global__ __launch_bounds__(256) void norm_kernel(const float* __restrict__ Wf,
                                                   float* __restrict__ wsInv) {
  __shared__ float rs[256];
  int tid = threadIdx.x;
  float q = 0.f;
  for (int e = tid; e < DD * DD; e += 256) { float v = Wf[e]; q += v * v; }
  rs[tid] = q;
  __syncthreads();
  for (int s = 128; s > 0; s >>= 1) {
    if (tid < s) rs[tid] += rs[tid + s];
    __syncthreads();
  }
  if (tid == 0) wsInv[0] = rsqrtf(rs[0]);
}

// ---- K1: fused y = [x|act_state]@Wp^T + bp ; z = y@Wf*invNorm ; layernorm ---
__global__ __launch_bounds__(256) void state_kernel(
    const float* __restrict__ x, const float* __restrict__ post,
    const float* __restrict__ Wp, const float* __restrict__ bp,
    const float* __restrict__ Wf, const float* __restrict__ gsyn,
    const float* __restrict__ bsyn, const float* __restrict__ wsInv,
    float* __restrict__ stateOut) {
  __shared__ __align__(16) char smemRaw[50 * 1024];
  _Float16* sInp = (_Float16*)smemRaw;                // 16x1024 f16 (32KB)
  _Float16* sY   = (_Float16*)(smemRaw + 32 * 1024);  // 16x512  f16 (16KB)
  float*    sZ   = (float*)smemRaw;                   // 16x512  f32 (reuses sInp)
  float*    sSum = (float*)(smemRaw + 48 * 1024);     // 256
  float*    sSq  = sSum + 256;                        // 256

  int tid = threadIdx.x, lane = tid & 31, w = tid >> 5;
  int b0 = blockIdx.x * 16;

  // stage inp = concat(x, post[:,:,M-1]) as f16
  for (int e = tid; e < 16 * 1024; e += 256) {
    int r = e >> 10, k = e & 1023;
    float v = (k < DD)
                  ? x[(size_t)(b0 + r) * DD + k]
                  : post[(size_t)(b0 + r) * (DD * MM) + (size_t)(k - DD) * MM + (MM - 1)];
    sInp[e] = (_Float16)v;
  }
  __syncthreads();

  // phase 1: y (16x512) = inp @ Wp^T + bp, K=1024.  wave w -> ntiles [4w..4w+3]
  for (int j = 0; j < 4; ++j) {
    int nt = w * 4 + j;
    int n = nt * 16 + (lane & 15);
    v8f acc = {};
    const float* wr = Wp + (size_t)n * 1024;  // B[k][n] = Wp[n][k]
    for (int k0 = 0; k0 < 1024; k0 += 32) {
      v16h a = frag_a_lds(sInp, 1024, 0, k0, lane);
      v16h bf;
      int kb = k0 + ((lane & 16) ? 16 : 0);
#pragma unroll
      for (int i = 0; i < 16; ++i) bf[i] = (_Float16)wr[kb + i];
      acc = wmma_f16(a, bf, acc);
    }
    float bias = bp[n];
    int rb = (lane & 16) ? 8 : 0;
#pragma unroll
    for (int r = 0; r < 8; ++r) sY[(rb + r) * DD + n] = (_Float16)(acc[r] + bias);
  }
  __syncthreads();

  // phase 2: z (16x512) = y @ Wf * invNorm, K=512
  float invN = wsInv[0];
  for (int j = 0; j < 4; ++j) {
    int nt = w * 4 + j;
    int n = nt * 16 + (lane & 15);
    v8f acc = {};
    for (int k0 = 0; k0 < DD; k0 += 32) {
      v16h a = frag_a_lds(sY, DD, 0, k0, lane);
      v16h bf;
      int kb = k0 + ((lane & 16) ? 16 : 0);
#pragma unroll
      for (int i = 0; i < 16; ++i) bf[i] = (_Float16)Wf[(size_t)(kb + i) * DD + n];
      acc = wmma_f16(a, bf, acc);
    }
    int rb = (lane & 16) ? 8 : 0;
#pragma unroll
    for (int r = 0; r < 8; ++r) sZ[(rb + r) * DD + n] = acc[r] * invN;
  }
  __syncthreads();

  // phase 3: layernorm over D=512 per row (16 threads per row, 32 elems each)
  int row = tid >> 4, sub = tid & 15;
  float s = 0.f, q = 0.f;
  for (int c = sub * 32; c < sub * 32 + 32; ++c) {
    float v = sZ[row * DD + c];
    s += v; q += v * v;
  }
  sSum[tid] = s; sSq[tid] = q;
  __syncthreads();
  if (sub == 0) {
    float S = 0.f, Q = 0.f;
    for (int t = 0; t < 16; ++t) { S += sSum[row * 16 + t]; Q += sSq[row * 16 + t]; }
    float mean = S * (1.0f / DD);
    float var = Q * (1.0f / DD) - mean * mean;
    sSum[row * 16] = mean;
    sSq[row * 16] = rsqrtf(var + 1e-5f);
  }
  __syncthreads();
  float mean = sSum[row * 16], rstd = sSq[row * 16];
  for (int c = sub * 32; c < sub * 32 + 32; ++c) {
    float v = sZ[row * DD + c];
    stateOut[(size_t)(b0 + row) * DD + c] = (v - mean) * rstd * gsyn[c] + bsyn[c];
  }
}

// ---- K2: per-d NLM: h=st@w1a_d+b1a ; silu ; LN(H) ; o=h2@w1b_d ; act_new ----
__global__ __launch_bounds__(256) void perd_kernel(
    const float* __restrict__ pre, const float* __restrict__ state,
    const float* __restrict__ w1a, const float* __restrict__ b1a,
    const float* __restrict__ gnlm, const float* __restrict__ bnlm,
    const float* __restrict__ w1b, const float* __restrict__ b1b,
    float* __restrict__ actOut) {
  __shared__ _Float16 sA[BB * MM];       // 64x32
  __shared__ _Float16 sB[MM * 2 * HH];   // 32x128
  __shared__ float    sH[BB * 2 * HH];   // 64x128
  __shared__ float    sB1a[2 * HH];
  __shared__ float    sW1b[2 * HH];      // [hh*2+c]
  __shared__ float    sG[HH], sBn[HH];

  int d = blockIdx.x;
  int tid = threadIdx.x, lane = tid & 31, w = tid >> 5;

  // stage st[:,d,:]: st[b][m] = m<31 ? pre[b][d][m+1] : state[b][d]
  for (int e = tid; e < BB * MM; e += 256) {
    int b = e >> 5, m = e & 31;
    float v = (m < MM - 1) ? pre[(size_t)b * (DD * MM) + (size_t)d * MM + m + 1]
                           : state[(size_t)b * DD + d];
    sA[e] = (_Float16)v;
  }
  // stage w1a[:, :, d] (32x128, stride-D gather, L2-amortized across blocks)
  for (int e = tid; e < MM * 2 * HH; e += 256) {
    int m = e >> 7, hc = e & 127;
    sB[e] = (_Float16)w1a[(size_t)m * (2 * HH * DD) + (size_t)hc * DD + d];
  }
  if (tid < 2 * HH) {
    sB1a[tid] = b1a[(size_t)d * (2 * HH) + tid];
    sW1b[tid] = w1b[(size_t)(tid >> 1) * (2 * DD) + (size_t)(tid & 1) * DD + d];
  }
  if (tid < HH) { sG[tid] = gnlm[tid]; sBn[tid] = bnlm[tid]; }
  __syncthreads();

  // h (64x128) = st @ w1a_d + b1a : 4 Mtiles x 8 Ntiles, single K=32 step
  int mt = w >> 1;
  for (int j = 0; j < 4; ++j) {
    int nt = (w & 1) * 4 + j;
    v16h a = frag_a_lds(sA, MM, mt * 16, 0, lane);
    v16h bf = frag_b_lds(sB, 2 * HH, 0, nt * 16, lane);
    v8f acc = {};
    acc = wmma_f16(a, bf, acc);
    int col = nt * 16 + (lane & 15);
    int rb = mt * 16 + ((lane & 16) ? 8 : 0);
    float bias = sB1a[col];
#pragma unroll
    for (int r = 0; r < 8; ++r) sH[(rb + r) * (2 * HH) + col] = acc[r] + bias;
  }
  __syncthreads();

  // silu gate in place: h2[b][j] = a * sigmoid(g)
  for (int e = tid; e < BB * HH; e += 256) {
    int b = e >> 6, jj = e & 63;
    float av = sH[b * (2 * HH) + jj];
    float gv = sH[b * (2 * HH) + HH + jj];
    sH[b * (2 * HH) + jj] = av * (1.0f / (1.0f + __expf(-gv)));
  }
  __syncthreads();

  // per-row LN(H=64) then 64x2 dot + final gate; one thread per batch row
  if (tid < BB) {
    int b = tid;
    float S = 0.f, Q = 0.f;
    for (int jj = 0; jj < HH; ++jj) { float v = sH[b * (2 * HH) + jj]; S += v; Q += v * v; }
    float mean = S * (1.0f / HH);
    float var = Q * (1.0f / HH) - mean * mean;
    float rstd = rsqrtf(var + 1e-5f);
    float o0 = b1b[(size_t)d * 2 + 0];
    float o1 = b1b[(size_t)d * 2 + 1];
    for (int hh = 0; hh < HH; ++hh) {
      float v = (sH[b * (2 * HH) + hh] - mean) * rstd * sG[hh] + sBn[hh];
      o0 += v * sW1b[hh * 2 + 0];
      o1 += v * sW1b[hh * 2 + 1];
    }
    actOut[(size_t)b * DD + d] = o0 * (1.0f / (1.0f + __expf(-o1)));
  }
}

// ---- K3: sync pass1: v = (dw^2*alpha + outer(an,an)) * rsqrt(dw^2*beta+1) ---
#define NBLK 256  // blocks per batch row; 1024 elems/block, 4/thread
__global__ __launch_bounds__(256) void sync_pass1(
    const float* __restrict__ dpar, const float* __restrict__ alpha,
    const float* __restrict__ beta, const float* __restrict__ act,
    float* __restrict__ out, float* __restrict__ part) {
  __shared__ float sAn[DD];
  __shared__ float rs[256], rq[256];
  int tid = threadIdx.x, blk = blockIdx.x, b = blockIdx.y;
  sAn[tid]       = act[(size_t)b * DD + tid];
  sAn[tid + 256] = act[(size_t)b * DD + 256 + tid];
  __syncthreads();

  int i = blk * 1024 + tid * 4;
  size_t gi = (size_t)b * DSQ + i;
  // stream alpha/beta nontemporally: keep L2 free for the 64MB v buffer
  f32x4 al = __builtin_nontemporal_load((const f32x4*)(alpha + gi));
  f32x4 be = __builtin_nontemporal_load((const f32x4*)(beta + gi));
  f32x4 pp = *(const f32x4*)(dpar + i);  // 1MB, L2-resident across all b
  float ar = sAn[i >> 9];
  int c = i & 511;
  f32x4 v;
  float sum = 0.f, sq = 0.f;
#pragma unroll
  for (int u = 0; u < 4; ++u) {
    float p = fminf(fmaxf(pp[u], 0.0f), 15.0f);
    float dw = __expf(-p);
    float d2 = dw * dw;
    float pv = ar * sAn[c + u];
    float vv = (d2 * al[u] + pv) * rsqrtf(d2 * be[u] + 1.0f);
    v[u] = vv; sum += vv; sq += vv * vv;
  }
  *(f32x4*)(out + gi) = v;  // regular store: want this resident in 192MB L2
  rs[tid] = sum; rq[tid] = sq;
  __syncthreads();
  for (int s = 128; s > 0; s >>= 1) {
    if (tid < s) { rs[tid] += rs[tid + s]; rq[tid] += rq[tid + s]; }
    __syncthreads();
  }
  if (tid == 0) {
    part[((size_t)b * NBLK + blk) * 2 + 0] = rs[0];
    part[((size_t)b * NBLK + blk) * 2 + 1] = rq[0];
  }
}

// ---- K4: finalize mean / rstd per batch row ---------------------------------
__global__ __launch_bounds__(256) void sync_stats(const float* __restrict__ part,
                                                  float* __restrict__ stats) {
  __shared__ float rs[256], rq[256];
  int tid = threadIdx.x, b = blockIdx.x;
  rs[tid] = part[((size_t)b * NBLK + tid) * 2 + 0];
  rq[tid] = part[((size_t)b * NBLK + tid) * 2 + 1];
  __syncthreads();
  for (int s = 128; s > 0; s >>= 1) {
    if (tid < s) { rs[tid] += rs[tid + s]; rq[tid] += rq[tid + s]; }
    __syncthreads();
  }
  if (tid == 0) {
    float mean = rs[0] * (1.0f / (float)DSQ);
    float var = rq[0] * (1.0f / (float)DSQ) - mean * mean;
    stats[b * 2 + 0] = mean;
    stats[b * 2 + 1] = rsqrtf(var + 1e-5f);
  }
}

// ---- K5: sync pass2: normalize (v hits L2), nontemporal final store ---------
__global__ __launch_bounds__(256) void sync_pass2(
    const float* __restrict__ g, const float* __restrict__ bb,
    const float* __restrict__ stats, float* __restrict__ out) {
  int tid = threadIdx.x, blk = blockIdx.x, b = blockIdx.y;
  int i = blk * 1024 + tid * 4;
  size_t gi = (size_t)b * DSQ + i;
  float mean = stats[b * 2 + 0], rstd = stats[b * 2 + 1];
  f32x4 v = *(const f32x4*)(out + gi);
  f32x4 gg = *(const f32x4*)(g + i);
  f32x4 bv = *(const f32x4*)(bb + i);
  f32x4 r;
#pragma unroll
  for (int u = 0; u < 4; ++u) r[u] = (v[u] - mean) * rstd * gg[u] + bv[u];
  __builtin_nontemporal_store(r, (f32x4*)(out + gi));
}

extern "C" void kernel_launch(void* const* d_in, const int* in_sizes, int n_in,
                              void* d_out, int out_size, void* d_ws, size_t ws_size,
                              hipStream_t stream) {
  const float* x     = (const float*)d_in[0];
  const float* pre   = (const float*)d_in[1];
  const float* post  = (const float*)d_in[2];
  const float* alpha = (const float*)d_in[3];
  const float* beta  = (const float*)d_in[4];
  const float* Wp    = (const float*)d_in[5];
  const float* bp    = (const float*)d_in[6];
  const float* Wf    = (const float*)d_in[7];
  const float* gsyn  = (const float*)d_in[8];
  const float* bsyn  = (const float*)d_in[9];
  const float* w1a   = (const float*)d_in[10];
  const float* b1a   = (const float*)d_in[11];
  const float* gnlm  = (const float*)d_in[12];
  const float* bnlm  = (const float*)d_in[13];
  const float* w1b   = (const float*)d_in[14];
  const float* b1b   = (const float*)d_in[15];
  const float* dpar  = (const float*)d_in[16];
  const float* gsync = (const float*)d_in[17];
  const float* bsync = (const float*)d_in[18];
  float* out = (float*)d_out;

  float* ws       = (float*)d_ws;
  float* wsInv    = ws;                       // 1
  float* wsState  = ws + 16;                  // 64*512
  float* wsAct    = wsState + BB * DD;        // 64*512
  float* wsPart   = wsAct + BB * DD;          // 64*256*2
  float* wsStats  = wsPart + BB * NBLK * 2;   // 128

  norm_kernel<<<1, 256, 0, stream>>>(Wf, wsInv);
  state_kernel<<<4, 256, 0, stream>>>(x, post, Wp, bp, Wf, gsyn, bsyn, wsInv, wsState);
  perd_kernel<<<DD, 256, 0, stream>>>(pre, wsState, w1a, b1a, gnlm, bnlm, w1b, b1b, wsAct);
  dim3 gSync(NBLK, BB);
  sync_pass1<<<gSync, 256, 0, stream>>>(dpar, alpha, beta, wsAct, out, wsPart);
  sync_stats<<<BB, 256, 0, stream>>>(wsPart, wsStats);
  sync_pass2<<<gSync, 256, 0, stream>>>(gsync, bsync, wsStats, out);
}
